// MultiHeadAttention_38611755991513
// MI455X (gfx1250) — compile-verified, matrix operand order checked by reading, not run on hardware
//
#include <hip/hip_runtime.h>
#include <hip/hip_bf16.h>
#include <stdint.h>

// MI455X / gfx1250, wave32. Matmuls via v_wmma_f32_16x16x32_bf16; both the
// projection GEMMs and the attention K/V streaming stage tiles into LDS with
// the Tensor Data Mover (tensor_load_to_lds, TENSORcnt-synchronized, with
// pad_enable for bank-conflict-free fragment reads).
// B=2, D=1024, L=2048, H=16, DK=64.
#define B_  2
#define D_  1024
#define L_  2048
#define H_  16
#define DK_ 64

typedef __attribute__((ext_vector_type(16))) __bf16        v16bf;
typedef __attribute__((ext_vector_type(8)))  float         v8f;
typedef __attribute__((ext_vector_type(4)))  unsigned int  v4u;
typedef __attribute__((ext_vector_type(8)))  int           v8i;
typedef __attribute__((ext_vector_type(4)))  int           v4i;

struct U32x8 { uint4 a, b; };

// Build a 16-element bf16 fragment from two contiguous 16-byte chunks.
// CDNA5 16-bit A/B fragment layout: per lane, elements 0..7 <- k = (lane/16)*8 + 0..7,
// elements 8..15 <- k = 16 + (lane/16)*8 + 0..7.
__device__ __forceinline__ v16bf frag_from(const unsigned short* p0, const unsigned short* p1) {
  U32x8 u;
  u.a = *(const uint4*)p0;
  u.b = *(const uint4*)p1;
  return __builtin_bit_cast(v16bf, u);
}

__device__ __forceinline__ v8f wmma_bf16(v16bf a, v16bf b, v8f c) {
  return __builtin_amdgcn_wmma_f32_16x16x32_bf16(false, a, false, b, (short)0, c, false, false);
}

__device__ __forceinline__ unsigned short f2bf(float f) {
  unsigned int u = __builtin_bit_cast(unsigned int, f);
  u += 0x7fffu + ((u >> 16) & 1u);   // round-to-nearest-even
  return (unsigned short)(u >> 16);
}

// ---------------------------------------------------------------------------
// Tensor Data Mover: async 2-D bf16 tile load (global -> LDS), with optional
// LDS destination padding for bank-conflict-free fragment reads.
// D# bitfields per CDNA5 ISA.
//   pad_interval code c -> pad after 2^(c+1) DWORDs; pad_amount code a -> (a+1) DWORDs.
// ---------------------------------------------------------------------------
__device__ __forceinline__ void tdm_load_2d_bf16(unsigned lds_off, const void* gaddr,
                                                 unsigned tensor_d0, unsigned tensor_d1,
                                                 unsigned stride0,
                                                 unsigned tile_d0, unsigned tile_d1,
                                                 unsigned pad_interval_code,
                                                 unsigned pad_amount_code) {
  unsigned long long ga = (unsigned long long)gaddr;
  v4u g0;
  g0[0] = 1u;                                           // count=1 (user mode), no gather
  g0[1] = lds_off;                                      // lds_addr (bytes)
  g0[2] = (unsigned)(ga & 0xffffffffu);                 // global_addr[31:0]
  g0[3] = (unsigned)((ga >> 32) & 0x01ffffffu)          // global_addr[56:32]
        | (2u << 30);                                   // type = 2 ("image")
  unsigned pad_en = (pad_amount_code | pad_interval_code) ? 1u : 0u;
  v8i g1;
  g1[0] = (int)((1u << 16)                              // data_size = 1 -> 2 bytes
        | (pad_en << 20)
        | (pad_interval_code << 22)
        | (pad_amount_code << 25));
  g1[1] = (int)((tensor_d0 & 0xffffu) << 16);           // tensor_dim0[15:0] @ bits 63:48
  g1[2] = (int)(((tensor_d0 >> 16) & 0xffffu)           // tensor_dim0[31:16]
        | ((tensor_d1 & 0xffffu) << 16));               // tensor_dim1[15:0]
  g1[3] = (int)(((tensor_d1 >> 16) & 0xffffu)           // tensor_dim1[31:16]
        | ((tile_d0 & 0xffffu) << 16));                 // tile_dim0
  g1[4] = (int)(tile_d1 & 0xffffu);                     // tile_dim1 (tile_dim2 = 0)
  g1[5] = (int)stride0;                                 // tensor_dim0_stride[31:0]
  g1[6] = 0;                                            // stride0[47:32]=0, dim1_stride lo=0
  g1[7] = 0;
  v4i z4 = {0, 0, 0, 0};
#if defined(__clang_major__) && (__clang_major__ >= 23)
  v8i z8 = {0, 0, 0, 0, 0, 0, 0, 0};
  __builtin_amdgcn_tensor_load_to_lds(g0, g1, z4, z4, z8, 0);
#else
  __builtin_amdgcn_tensor_load_to_lds(g0, g1, z4, z4, 0);
#endif
}

// ---------------------------------------------------------------------------
// fp32 -> bf16 elementwise (weights), vectorized x4
// ---------------------------------------------------------------------------
__global__ __launch_bounds__(256) void k_f32_to_bf16(const float* __restrict__ src,
                                                     unsigned short* __restrict__ dst, int n4) {
  int i = blockIdx.x * blockDim.x + threadIdx.x;
  if (i >= n4) return;
  float4 f = ((const float4*)src)[i];
  ushort4 o;
  o.x = f2bf(f.x); o.y = f2bf(f.y); o.z = f2bf(f.z); o.w = f2bf(f.w);
  ((ushort4*)dst)[i] = o;
}

// ---------------------------------------------------------------------------
// query [B, D, L] f32  ->  Xt [B, L, D] bf16  (LDS-tiled transpose)
// ---------------------------------------------------------------------------
__global__ __launch_bounds__(256) void k_transpose_bf16(const float* __restrict__ x,
                                                        unsigned short* __restrict__ xt) {
  __shared__ float tile[32][33];
  int b  = blockIdx.z;
  int l0 = blockIdx.x * 32;
  int d0 = blockIdx.y * 32;
  int tx = threadIdx.x, ty = threadIdx.y;   // block (32, 8)
  #pragma unroll
  for (int i = 0; i < 4; ++i) {
    int d = d0 + ty + i * 8;
    tile[ty + i * 8][tx] = x[((size_t)b * D_ + d) * L_ + l0 + tx];
  }
  __syncthreads();
  #pragma unroll
  for (int i = 0; i < 4; ++i) {
    int l = l0 + ty + i * 8;
    xt[((size_t)b * L_ + l) * D_ + d0 + tx] = f2bf(tile[tx][ty + i * 8]);
  }
}

// ---------------------------------------------------------------------------
// Projection GEMM: C[b][o][l] = sum_d W[o][d] * X[b][l][d]  (+bias)
// Block 256 thr = 8 waves, block tile 128(M) x 128(N), wave tile 32 x 64.
// W and X K-step tiles (128x32 bf16) are DMA'd to LDS by the TDM (wave 0,
// double-buffered, TENSORcnt + one barrier per K-step). Padded 20-DW rows
// make every 16-lane ds_load_b128 fragment read bank-conflict-free.
// MODE 0: Q -> [bh][l][dk] * 0.125   MODE 1: K -> [bh][l][dk]
// MODE 2: V -> [bh][dk][l] (transposed)   MODE 3: final f32 out [b][o][l]
// ---------------------------------------------------------------------------
#define GROW 40   // 32 data + 8 pad (bf16 elements) per staged GEMM tile row

template<int MODE>
__global__ __launch_bounds__(256) void k_proj(const unsigned short* __restrict__ W,
                                              const float* __restrict__ bias,
                                              const unsigned short* __restrict__ X,
                                              unsigned short* __restrict__ outH,
                                              float* __restrict__ outF) {
  __shared__ __align__(16) unsigned short Wl[2][128 * GROW];
  __shared__ __align__(16) unsigned short Xl[2][128 * GROW];

  const int b    = blockIdx.z;
  const int Mblk = blockIdx.y * 128;
  const int Nblk = blockIdx.x * 128;
  const int wave = threadIdx.x >> 5;
  const int lane = threadIdx.x & 31;
  const int col  = lane & 15;
  const int half = lane >> 4;
  const int wm   = wave >> 1;            // 0..3 -> 32-row M slice
  const int wn   = wave & 1;             // 0..1 -> 64-col N slice
  const int waveM = Mblk + wm * 32;
  const int waveN = Nblk + wn * 64;
  const unsigned short* Xb = X + (size_t)b * L_ * D_;

  const v8f vzero = {0.f, 0.f, 0.f, 0.f, 0.f, 0.f, 0.f, 0.f};
  v8f acc[2][4];
  #pragma unroll
  for (int i = 0; i < 2; ++i)
    #pragma unroll
    for (int j = 0; j < 4; ++j) acc[i][j] = vzero;

  // TDM prologue: K-step 0 into buffer 0.
  if (wave == 0) {
    tdm_load_2d_bf16((unsigned)(uintptr_t)&Wl[0][0], W + (size_t)Mblk * D_,
                     D_, D_, D_, 32, 128, /*pad: 16dw interval*/3, /*4dw*/3);
    tdm_load_2d_bf16((unsigned)(uintptr_t)&Xl[0][0], Xb + (size_t)Nblk * D_,
                     D_, L_, D_, 32, 128, 3, 3);
  }

  const int NK = D_ / 32;
  for (int it = 0; it < NK; ++it) {
    const int cur = it & 1;
    const int nxt = cur ^ 1;

    if (wave == 0) __builtin_amdgcn_s_wait_tensorcnt(0);  // step `it` landed
    __syncthreads();                                      // visible; buf[nxt] consumed
    if (wave == 0 && it + 1 < NK) {                       // prefetch step it+1 via TDM
      const int k1 = (it + 1) * 32;
      tdm_load_2d_bf16((unsigned)(uintptr_t)&Wl[nxt][0], W + (size_t)Mblk * D_ + k1,
                       D_, D_, D_, 32, 128, 3, 3);
      tdm_load_2d_bf16((unsigned)(uintptr_t)&Xl[nxt][0], Xb + (size_t)Nblk * D_ + k1,
                       D_, L_, D_, 32, 128, 3, 3);
    }

    const unsigned short* Wc = &Wl[cur][0];
    const unsigned short* Xc = &Xl[cur][0];

    v16bf af[2], bfr[4];
    #pragma unroll
    for (int i = 0; i < 2; ++i) {
      const unsigned short* p = Wc + (size_t)(wm * 32 + i * 16 + col) * GROW + half * 8;
      af[i] = frag_from(p, p + 16);
    }
    #pragma unroll
    for (int j = 0; j < 4; ++j) {
      const unsigned short* p = Xc + (size_t)(wn * 64 + j * 16 + col) * GROW + half * 8;
      bfr[j] = frag_from(p, p + 16);
    }
    #pragma unroll
    for (int i = 0; i < 2; ++i)
      #pragma unroll
      for (int j = 0; j < 4; ++j)
        acc[i][j] = wmma_bf16(af[i], bfr[j], acc[i][j]);
  }

  // Epilogue. C/D layout: m = r + 8*(lane/16), n = lane%16.
  #pragma unroll
  for (int i = 0; i < 2; ++i) {
    float bi[8];
    #pragma unroll
    for (int r = 0; r < 8; ++r) bi[r] = bias[waveM + i * 16 + r + 8 * half];
    #pragma unroll
    for (int j = 0; j < 4; ++j) {
      int n = waveN + j * 16 + col;                  // l
      #pragma unroll
      for (int r = 0; r < 8; ++r) {
        int o = waveM + i * 16 + r + 8 * half;       // out channel
        float v = acc[i][j][r] + bi[r];
        if constexpr (MODE == 0) {
          v *= 0.125f;                               // 1/sqrt(DK) after bias (matches ref)
          int h = o >> 6, dk = o & 63;
          outH[(((size_t)(b * H_ + h)) * L_ + n) * DK_ + dk] = f2bf(v);
        } else if constexpr (MODE == 1) {
          int h = o >> 6, dk = o & 63;
          outH[(((size_t)(b * H_ + h)) * L_ + n) * DK_ + dk] = f2bf(v);
        } else if constexpr (MODE == 2) {
          int h = o >> 6, dk = o & 63;
          outH[(((size_t)(b * H_ + h)) * DK_ + dk) * L_ + n] = f2bf(v);
        } else {
          outF[((size_t)b * D_ + o) * L_ + n] = v;
        }
      }
    }
  }
}

// ---------------------------------------------------------------------------
// Flash attention per (b,h). 8 waves/block, 16 query rows per wave.
// K/V chunks (32 keys) DMA'd to LDS by the TDM (wave 0 issues, double-buffered,
// TENSORcnt + one barrier per chunk). TDM pad_enable gives bank-conflict-free
// row strides: K rows 64+8 el (36 dw), V rows 32+8 el (20 dw).
// Mask is key-only; run_s==0 guard reproduces the all-masked -> 0 path.
// ---------------------------------------------------------------------------
#define KROW 72   // 64 data + 8 pad (bf16 elements)
#define VROW 40   // 32 data + 8 pad
#define PROW 40   // 32 data + 8 pad (hand-padded)

__global__ __launch_bounds__(256) void k_attention(const unsigned short* __restrict__ Qh,
                                                   const unsigned short* __restrict__ Kh,
                                                   const unsigned short* __restrict__ Vt,
                                                   const unsigned char* __restrict__ mask,
                                                   unsigned short* __restrict__ Yt) {
  __shared__ __align__(16) unsigned short Klds[2][32 * KROW];
  __shared__ __align__(16) unsigned short Vlds[2][64 * VROW];
  __shared__ __align__(16) unsigned short Pbuf[8][16 * PROW];

  const int bh   = blockIdx.y;                     // b*H + h
  const int b    = bh >> 4;
  const int h    = bh & 15;
  const int wave = threadIdx.x >> 5;
  const int lane = threadIdx.x & 31;
  const int col  = lane & 15;
  const int half = lane >> 4;
  const int q0   = blockIdx.x * 128 + wave * 16;
  unsigned short* Pw = &Pbuf[wave][0];

  const unsigned short* Qb = Qh + (size_t)bh * L_ * DK_;
  const unsigned short* Kb = Kh + (size_t)bh * L_ * DK_;
  const unsigned short* Vb = Vt + (size_t)bh * DK_ * L_;
  const unsigned char*  mb = mask + (size_t)b * L_;

  // Q A-fragments: row = q0+col, dk chunks [0,32) and [32,64)
  v16bf qf[2];
  #pragma unroll
  for (int c = 0; c < 2; ++c) {
    const unsigned short* p = Qb + (size_t)(q0 + col) * DK_ + c * 32 + half * 8;
    qf[c] = frag_from(p, p + 16);
  }

  const v8f vzero = {0.f, 0.f, 0.f, 0.f, 0.f, 0.f, 0.f, 0.f};
  v8f acc[4];
  #pragma unroll
  for (int t = 0; t < 4; ++t) acc[t] = vzero;
  float run_m[8], run_s[8];
  #pragma unroll
  for (int r = 0; r < 8; ++r) { run_m[r] = -1e30f; run_s[r] = 0.f; }

  // TDM prologue: chunk 0 into buffer 0 (wave 0 only; EXEC-independent DMA).
  if (wave == 0) {
    tdm_load_2d_bf16((unsigned)(uintptr_t)&Klds[0][0], Kb,
                     DK_, L_, DK_, DK_, 32, /*pad: 32dw interval*/4, /*4dw*/3);
    tdm_load_2d_bf16((unsigned)(uintptr_t)&Vlds[0][0], Vb,
                     L_, DK_, L_, 32, DK_, /*pad: 16dw interval*/3, /*4dw*/3);
  }

  const int NCHUNK = L_ / 32;
  for (int it = 0; it < NCHUNK; ++it) {
    const int n0  = it * 32;
    const int cur = it & 1;
    const int nxt = cur ^ 1;

    if (wave == 0) __builtin_amdgcn_s_wait_tensorcnt(0);  // chunk `it` landed
    __syncthreads();                                      // visible to all waves;
                                                          // also: buf[nxt] fully consumed
    if (wave == 0 && it + 1 < NCHUNK) {                   // prefetch chunk it+1 via TDM
      const int n1 = n0 + 32;
      tdm_load_2d_bf16((unsigned)(uintptr_t)&Klds[nxt][0], Kb + (size_t)n1 * DK_,
                       DK_, L_, DK_, DK_, 32, 4, 3);
      tdm_load_2d_bf16((unsigned)(uintptr_t)&Vlds[nxt][0], Vb + n1,
                       L_, DK_, L_, 32, DK_, 3, 3);
    }

    const unsigned short* Kc = &Klds[cur][0];
    const unsigned short* Vc = &Vlds[cur][0];

    // S = Q K^T for two 16-key subtiles (K fragments straight from LDS)
    v8f s[2];
    bool ok[2];
    #pragma unroll
    for (int st = 0; st < 2; ++st) {
      ok[st] = mb[n0 + st * 16 + col] != 0;
      const unsigned short* p0 = Kc + (size_t)(st * 16 + col) * KROW + half * 8;
      const unsigned short* p1 = Kc + (size_t)(st * 16 + col) * KROW + 32 + half * 8;
      v16bf kf0 = frag_from(p0, p0 + 16);
      v16bf kf1 = frag_from(p1, p1 + 16);
      s[st] = wmma_bf16(qf[1], kf1, wmma_bf16(qf[0], kf0, vzero));
    }

    // Online softmax over this 32-key chunk; emit P (bf16) into per-wave LDS.
    #pragma unroll
    for (int r = 0; r < 8; ++r) {
      float a0 = ok[0] ? s[0][r] : -1e30f;
      float a1 = ok[1] ? s[1][r] : -1e30f;
      float t = fmaxf(a0, a1);
      t = fmaxf(t, __shfl_xor(t, 1));
      t = fmaxf(t, __shfl_xor(t, 2));
      t = fmaxf(t, __shfl_xor(t, 4));
      t = fmaxf(t, __shfl_xor(t, 8));
      float nm = fmaxf(run_m[r], t);
      float p0 = ok[0] ? __expf(a0 - nm) : 0.0f;   // masked lanes contribute exactly 0
      float p1 = ok[1] ? __expf(a1 - nm) : 0.0f;
      float sc = __expf(run_m[r] - nm);
      float ps = p0 + p1;
      ps += __shfl_xor(ps, 1);
      ps += __shfl_xor(ps, 2);
      ps += __shfl_xor(ps, 4);
      ps += __shfl_xor(ps, 8);
      run_s[r] = run_s[r] * sc + ps;
      run_m[r] = nm;
      acc[0][r] *= sc; acc[1][r] *= sc; acc[2][r] *= sc; acc[3][r] *= sc;
      int row = r + 8 * half;
      Pw[row * PROW + col]      = f2bf(p0);
      Pw[row * PROW + 16 + col] = f2bf(p1);
    }
    // Per-wave LDS is processed in order (DScnt in-order within a wave), so the
    // fragment read below sees the stores above without a block barrier.

    const unsigned short* pp = Pw + (size_t)col * PROW + half * 8;
    v16bf pf = frag_from(pp, pp + 16);

    // O += P * V  (V tile in LDS as [dk][32], padded rows)
    #pragma unroll
    for (int t4 = 0; t4 < 4; ++t4) {
      const unsigned short* vp = Vc + (size_t)(t4 * 16 + col) * VROW + half * 8;
      v16bf vf = frag_from(vp, vp + 16);
      acc[t4] = wmma_bf16(pf, vf, acc[t4]);
    }
  }

  // Normalize and store combined-heads Y [B, L, D] bf16.
  #pragma unroll
  for (int r = 0; r < 8; ++r) {
    float inv = run_s[r] > 0.0f ? 1.0f / run_s[r] : 0.0f;
    int l = q0 + r + 8 * half;
    #pragma unroll
    for (int t4 = 0; t4 < 4; ++t4) {
      int dkg = h * 64 + t4 * 16 + col;
      Yt[((size_t)b * L_ + l) * D_ + dkg] = f2bf(acc[t4][r] * inv);
    }
  }
}

// ---------------------------------------------------------------------------
// Launcher
// ---------------------------------------------------------------------------
extern "C" void kernel_launch(void* const* d_in, const int* in_sizes, int n_in,
                              void* d_out, int out_size, void* d_ws, size_t ws_size,
                              hipStream_t stream) {
  const float*         x    = (const float*)d_in[0];          // [B, D, L]
  const unsigned char* mask = (const unsigned char*)d_in[1];  // jax bool -> 1 byte, [B,1,L]
  const float* Wq = (const float*)d_in[2];
  const float* bq = (const float*)d_in[3];
  const float* Wk = (const float*)d_in[4];
  const float* bk = (const float*)d_in[5];
  const float* Wv = (const float*)d_in[6];
  const float* bv = (const float*)d_in[7];
  const float* Wo = (const float*)d_in[8];
  const float* bo = (const float*)d_in[9];
  float* out = (float*)d_out;                                 // [B, D, L] f32

  // Workspace carve-out (total ~48 MB)
  char* ws = (char*)d_ws;
  size_t off = 0;
  auto take = [&](size_t bytes) -> char* {
    char* p = ws + off;
    off += (bytes + 255) & ~(size_t)255;
    return p;
  };
  unsigned short* Wqb = (unsigned short*)take((size_t)D_ * D_ * 2);
  unsigned short* Wkb = (unsigned short*)take((size_t)D_ * D_ * 2);
  unsigned short* Wvb = (unsigned short*)take((size_t)D_ * D_ * 2);
  unsigned short* Wob = (unsigned short*)take((size_t)D_ * D_ * 2);
  unsigned short* Xt  = (unsigned short*)take((size_t)B_ * L_ * D_ * 2);       // [B,L,D]
  unsigned short* Qh  = (unsigned short*)take((size_t)B_ * H_ * L_ * DK_ * 2); // [bh,l,dk]
  unsigned short* Kh  = (unsigned short*)take((size_t)B_ * H_ * L_ * DK_ * 2); // [bh,l,dk]
  unsigned short* Vt  = (unsigned short*)take((size_t)B_ * H_ * DK_ * L_ * 2); // [bh,dk,l]
  unsigned short* Yt  = (unsigned short*)take((size_t)B_ * L_ * D_ * 2);       // [B,L,D]

  // 1) convert weights to bf16
  int n4 = D_ * D_ / 4;
  dim3 cb(256), cg((n4 + 255) / 256);
  k_f32_to_bf16<<<cg, cb, 0, stream>>>(Wq, Wqb, n4);
  k_f32_to_bf16<<<cg, cb, 0, stream>>>(Wk, Wkb, n4);
  k_f32_to_bf16<<<cg, cb, 0, stream>>>(Wv, Wvb, n4);
  k_f32_to_bf16<<<cg, cb, 0, stream>>>(Wo, Wob, n4);

  // 2) transpose+convert activations: [B,D,L] f32 -> [B,L,D] bf16
  dim3 tb(32, 8), tg(L_ / 32, D_ / 32, B_);
  k_transpose_bf16<<<tg, tb, 0, stream>>>(x, Xt);

  // 3) Q/K/V projections (TDM-staged WMMA GEMMs), writing attention-ready layouts
  dim3 pb(256), pg(L_ / 128, D_ / 128, B_);
  k_proj<0><<<pg, pb, 0, stream>>>(Wqb, bq, Xt, Qh, nullptr);
  k_proj<1><<<pg, pb, 0, stream>>>(Wkb, bk, Xt, Kh, nullptr);
  k_proj<2><<<pg, pb, 0, stream>>>(Wvb, bv, Xt, Vt, nullptr);

  // 4) flash attention (TDM-staged K/V)
  dim3 ab(256), ag(L_ / 128, B_ * H_);
  k_attention<<<ag, ab, 0, stream>>>(Qh, Kh, Vt, mask, Yt);

  // 5) output projection (f32 result)
  k_proj<3><<<pg, pb, 0, stream>>>(Wob, bo, Yt, nullptr, out);
}
